// Attn_bahdanau_47390669144872
// MI455X (gfx1250) — matrix operand order repaired and checked by reading, not compiled, not run on hardware
//
#include <hip/hip_runtime.h>

// ---------------------------------------------------------------------------
// Attention: context = softmax(Q @ K^T) @ V, also outputs attn = softmax(...)
// B=8, Lq=Lk=2048, D=1024, fp32 in / fp32 out.
// Fast path (needs 128MiB ws): f32->bf16 convert pass, then bf16 WMMA GEMMs
//   whose inner loops are pure copy + wmma (no conversion VALU), K-step 64.
// Fallback path: round-2 kernels converting on the fly (K-step 32).
// ---------------------------------------------------------------------------

typedef __attribute__((ext_vector_type(16))) __bf16 v16bf;
typedef __attribute__((ext_vector_type(8)))  __bf16 v8bf;
typedef __attribute__((ext_vector_type(8)))  float  v8f;

#define TB   256   // 8 waves (wave32)
#define BM   128   // block tile M (4 wave-rows x 32)
#define BN   64    // block tile N (2 wave-cols x 32)
#define BK   32    // fallback K step
#define LDP  40    // fallback LDS pitch (elements)
#define FBK  64    // fast-path K step (2 wmma-K per staging step)
#define FLDP 72    // fast-path LDS pitch (144B rows; 16B chunks aligned)

__device__ __forceinline__ unsigned short bf16rne(float f) {
  union { float f; unsigned u; } in; in.f = f;
  unsigned u = in.u;
  unsigned r = u + 0x7FFFu + ((u >> 16) & 1u);
  return (unsigned short)(r >> 16);
}

__device__ __forceinline__ __bf16 f2bf(float f) {
  union { unsigned short s; __bf16 b; } out; out.s = bf16rne(f);
  return out.b;
}

__device__ __forceinline__ void st4(__bf16* d, float4 v) {
  d[0] = f2bf(v.x); d[1] = f2bf(v.y); d[2] = f2bf(v.z); d[3] = f2bf(v.w);
}

union Frag { v16bf v; v8bf h[2]; };

#define WMMA_BF16(A, B, C) \
  __builtin_amdgcn_wmma_f32_16x16x32_bf16(false, (A), false, (B), (short)0, (C), false, false)

// ===========================================================================
// FAST PATH
// ===========================================================================

// -------- f32 -> bf16 bulk convert (grid-stride, float4 in / uint2 out) ----
__global__ __launch_bounds__(TB)
void f32_to_bf16_kernel(const float* __restrict__ src, __bf16* __restrict__ dst,
                        long long n4) {
  long long i = (long long)blockIdx.x * blockDim.x + threadIdx.x;
  const long long stride = (long long)gridDim.x * blockDim.x;
  for (; i < n4; i += stride) {
    float4 v = ((const float4*)src)[i];
    uint2 o;
    o.x = (unsigned)bf16rne(v.x) | ((unsigned)bf16rne(v.y) << 16);
    o.y = (unsigned)bf16rne(v.z) | ((unsigned)bf16rne(v.w) << 16);
    ((uint2*)dst)[i] = o;
  }
}

// -------- GEMM1 fast: S = Qb @ Kb^T  (bf16 operands from global) -----------
__global__ __launch_bounds__(TB)
void qk_scores_bf_kernel(const __bf16* __restrict__ Qb, const __bf16* __restrict__ Kb,
                         float* __restrict__ S, int Lq, int Lk, int D) {
  __shared__ __bf16 Qs[2][BM][FLDP];   // [buf][m][d]
  __shared__ __bf16 Ks[2][BN][FLDP];   // [buf][k][d]

  const int b    = blockIdx.z;
  const int m0   = blockIdx.y * BM;
  const int n0   = blockIdx.x * BN;
  const int t    = threadIdx.x;
  const int lane = t & 31, wid = t >> 5;
  const int wm   = wid >> 1, wn = wid & 1;
  const int half = lane >> 4, l16 = lane & 15;

  const __bf16* Ag = Qb + ((size_t)b * Lq + m0) * D;
  const __bf16* Bg = Kb + ((size_t)b * Lk + n0) * D;

  // A 128x64 -> thread: row t>>1, 32 cols; B 64x64 -> thread: row t>>2, 16 cols
  const int ar = t >> 1, ac0 = (t & 1) * 32;
  const int br = t >> 2, bc0 = (t & 3) * 16;
  const __bf16* pa = Ag + (size_t)ar * D + ac0;
  const __bf16* pb = Bg + (size_t)br * D + bc0;

  v8f c00 = {}, c01 = {}, c10 = {}, c11 = {};

  v8bf ra0, ra1, ra2, ra3, rb0, rb1;
  ra0 = *(const v8bf*)(pa + 0);  ra1 = *(const v8bf*)(pa + 8);
  ra2 = *(const v8bf*)(pa + 16); ra3 = *(const v8bf*)(pa + 24);
  rb0 = *(const v8bf*)(pb + 0);  rb1 = *(const v8bf*)(pb + 8);
  *(v8bf*)&Qs[0][ar][ac0 + 0]  = ra0; *(v8bf*)&Qs[0][ar][ac0 + 8]  = ra1;
  *(v8bf*)&Qs[0][ar][ac0 + 16] = ra2; *(v8bf*)&Qs[0][ar][ac0 + 24] = ra3;
  *(v8bf*)&Ks[0][br][bc0 + 0]  = rb0; *(v8bf*)&Ks[0][br][bc0 + 8]  = rb1;
  __syncthreads();

  const int NIT = D / FBK;
  for (int it = 0; it < NIT; ++it) {
    const int cur = it & 1, nxt = cur ^ 1;
    const bool has = (it + 1 < NIT);
    if (has) {
      const int k0n = (it + 1) * FBK;
      ra0 = *(const v8bf*)(pa + k0n + 0);  ra1 = *(const v8bf*)(pa + k0n + 8);
      ra2 = *(const v8bf*)(pa + k0n + 16); ra3 = *(const v8bf*)(pa + k0n + 24);
      rb0 = *(const v8bf*)(pb + k0n + 0);  rb1 = *(const v8bf*)(pb + k0n + 8);
      if (it + 2 < NIT) {
        __builtin_prefetch(pa + (it + 2) * FBK, 0, 1);
        __builtin_prefetch(pb + (it + 2) * FBK, 0, 1);
      }
    }

    const __bf16* qa0 = &Qs[cur][wm * 32 + l16][0];
    const __bf16* qa1 = &Qs[cur][wm * 32 + 16 + l16][0];
    const __bf16* kb0 = &Ks[cur][wn * 32 + l16][0];
    const __bf16* kb1 = &Ks[cur][wn * 32 + 16 + l16][0];
#pragma unroll
    for (int kk = 0; kk < FBK; kk += 32) {
      Frag A0, A1, B0, B1;
      A0.h[0] = *(const v8bf*)(qa0 + kk + 16 * half);
      A0.h[1] = *(const v8bf*)(qa0 + kk + 16 * half + 8);
      A1.h[0] = *(const v8bf*)(qa1 + kk + 16 * half);
      A1.h[1] = *(const v8bf*)(qa1 + kk + 16 * half + 8);
      B0.h[0] = *(const v8bf*)(kb0 + kk + 16 * half);
      B0.h[1] = *(const v8bf*)(kb0 + kk + 16 * half + 8);
      B1.h[0] = *(const v8bf*)(kb1 + kk + 16 * half);
      B1.h[1] = *(const v8bf*)(kb1 + kk + 16 * half + 8);
      c00 = WMMA_BF16(A0.v, B0.v, c00);
      c01 = WMMA_BF16(A0.v, B1.v, c01);
      c10 = WMMA_BF16(A1.v, B0.v, c10);
      c11 = WMMA_BF16(A1.v, B1.v, c11);
    }

    if (has) {
      *(v8bf*)&Qs[nxt][ar][ac0 + 0]  = ra0; *(v8bf*)&Qs[nxt][ar][ac0 + 8]  = ra1;
      *(v8bf*)&Qs[nxt][ar][ac0 + 16] = ra2; *(v8bf*)&Qs[nxt][ar][ac0 + 24] = ra3;
      *(v8bf*)&Ks[nxt][br][bc0 + 0]  = rb0; *(v8bf*)&Ks[nxt][br][bc0 + 8]  = rb1;
    }
    __syncthreads();
  }

  float* Sg = S + ((size_t)b * Lq + m0 + wm * 32) * Lk + n0 + wn * 32;
#pragma unroll
  for (int r = 0; r < 8; ++r) {
    const int row = r + 8 * half;
    Sg[(size_t)row * Lk + l16]             = c00[r];
    Sg[(size_t)row * Lk + 16 + l16]        = c01[r];
    Sg[(size_t)(row + 16) * Lk + l16]      = c10[r];
    Sg[(size_t)(row + 16) * Lk + 16 + l16] = c11[r];
  }
}

// -------- softmax that also emits a bf16 copy of the result ----------------
__global__ __launch_bounds__(TB)
void softmax_bf_kernel(float* __restrict__ attn, __bf16* __restrict__ attn_bf, int Lk) {
  __shared__ float red[TB];
  float*  p  = attn    + (size_t)blockIdx.x * Lk;
  __bf16* pb = attn_bf + (size_t)blockIdx.x * Lk;
  const int t = threadIdx.x;

  float m = -3.402823466e+38f;
  for (int i = t; i < Lk; i += TB) m = fmaxf(m, p[i]);
  red[t] = m; __syncthreads();
  for (int s = TB / 2; s > 0; s >>= 1) {
    if (t < s) red[t] = fmaxf(red[t], red[t + s]);
    __syncthreads();
  }
  m = red[0]; __syncthreads();

  float sum = 0.0f;
  for (int i = t; i < Lk; i += TB) {
    float e = __expf(p[i] - m);
    p[i] = e;
    sum += e;
  }
  red[t] = sum; __syncthreads();
  for (int s = TB / 2; s > 0; s >>= 1) {
    if (t < s) red[t] += red[t + s];
    __syncthreads();
  }
  const float inv = 1.0f / red[0];
  for (int i = t; i < Lk; i += TB) {
    const float v = p[i] * inv;
    p[i]  = v;
    pb[i] = f2bf(v);
  }
}

// -------- GEMM2 fast: O = Pb @ Vb  (bf16 operands from global) -------------
__global__ __launch_bounds__(TB)
void attn_v_bf_kernel(const __bf16* __restrict__ Pb, const __bf16* __restrict__ Vb,
                      float* __restrict__ O, int Lq, int Lk, int D) {
  __shared__ __bf16 As[2][BM][FLDP];    // [buf][m][k]
  __shared__ __bf16 VTs[2][BN][FLDP];   // [buf][d][k] (transposed V tile)

  const int b    = blockIdx.z;
  const int m0   = blockIdx.y * BM;
  const int n0   = blockIdx.x * BN;     // n = d dimension
  const int t    = threadIdx.x;
  const int lane = t & 31, wid = t >> 5;
  const int wm   = wid >> 1, wn = wid & 1;
  const int half = lane >> 4, l16 = lane & 15;

  const __bf16* Pg = Pb + ((size_t)b * Lq + m0) * Lk;
  const __bf16* Vg = Vb + (size_t)b * Lk * D;

  // A 128x64 -> thread: row t>>1, 32 cols; V 64(k) x 64(d) -> k t>>2, 16 d
  const int ar  = t >> 1, ac0 = (t & 1) * 32;
  const int vkk = t >> 2, vd0 = (t & 3) * 16;
  const __bf16* pa = Pg + (size_t)ar * Lk + ac0;

  v8f c00 = {}, c01 = {}, c10 = {}, c11 = {};

  v8bf ra0, ra1, ra2, ra3, rv0, rv1;
  ra0 = *(const v8bf*)(pa + 0);  ra1 = *(const v8bf*)(pa + 8);
  ra2 = *(const v8bf*)(pa + 16); ra3 = *(const v8bf*)(pa + 24);
  rv0 = *(const v8bf*)(Vg + (size_t)vkk * D + n0 + vd0);
  rv1 = *(const v8bf*)(Vg + (size_t)vkk * D + n0 + vd0 + 8);
  *(v8bf*)&As[0][ar][ac0 + 0]  = ra0; *(v8bf*)&As[0][ar][ac0 + 8]  = ra1;
  *(v8bf*)&As[0][ar][ac0 + 16] = ra2; *(v8bf*)&As[0][ar][ac0 + 24] = ra3;
#pragma unroll
  for (int j = 0; j < 8; ++j) {
    VTs[0][vd0 + j][vkk]     = rv0[j];
    VTs[0][vd0 + 8 + j][vkk] = rv1[j];
  }
  __syncthreads();

  const int NIT = Lk / FBK;
  for (int it = 0; it < NIT; ++it) {
    const int cur = it & 1, nxt = cur ^ 1;
    const bool has = (it + 1 < NIT);
    if (has) {
      const int k0n = (it + 1) * FBK;
      ra0 = *(const v8bf*)(pa + k0n + 0);  ra1 = *(const v8bf*)(pa + k0n + 8);
      ra2 = *(const v8bf*)(pa + k0n + 16); ra3 = *(const v8bf*)(pa + k0n + 24);
      rv0 = *(const v8bf*)(Vg + (size_t)(k0n + vkk) * D + n0 + vd0);
      rv1 = *(const v8bf*)(Vg + (size_t)(k0n + vkk) * D + n0 + vd0 + 8);
      if (it + 2 < NIT) {
        __builtin_prefetch(pa + (it + 2) * FBK, 0, 1);
        __builtin_prefetch(Vg + (size_t)((it + 2) * FBK + vkk) * D + n0 + vd0, 0, 1);
      }
    }

    const __bf16* pa0 = &As[cur][wm * 32 + l16][0];
    const __bf16* pa1 = &As[cur][wm * 32 + 16 + l16][0];
    const __bf16* pb0 = &VTs[cur][wn * 32 + l16][0];
    const __bf16* pb1 = &VTs[cur][wn * 32 + 16 + l16][0];
#pragma unroll
    for (int kk = 0; kk < FBK; kk += 32) {
      Frag A0, A1, B0, B1;
      A0.h[0] = *(const v8bf*)(pa0 + kk + 16 * half);
      A0.h[1] = *(const v8bf*)(pa0 + kk + 16 * half + 8);
      A1.h[0] = *(const v8bf*)(pa1 + kk + 16 * half);
      A1.h[1] = *(const v8bf*)(pa1 + kk + 16 * half + 8);
      B0.h[0] = *(const v8bf*)(pb0 + kk + 16 * half);
      B0.h[1] = *(const v8bf*)(pb0 + kk + 16 * half + 8);
      B1.h[0] = *(const v8bf*)(pb1 + kk + 16 * half);
      B1.h[1] = *(const v8bf*)(pb1 + kk + 16 * half + 8);
      c00 = WMMA_BF16(A0.v, B0.v, c00);
      c01 = WMMA_BF16(A0.v, B1.v, c01);
      c10 = WMMA_BF16(A1.v, B0.v, c10);
      c11 = WMMA_BF16(A1.v, B1.v, c11);
    }

    if (has) {
      *(v8bf*)&As[nxt][ar][ac0 + 0]  = ra0; *(v8bf*)&As[nxt][ar][ac0 + 8]  = ra1;
      *(v8bf*)&As[nxt][ar][ac0 + 16] = ra2; *(v8bf*)&As[nxt][ar][ac0 + 24] = ra3;
#pragma unroll
      for (int j = 0; j < 8; ++j) {
        VTs[nxt][vd0 + j][vkk]     = rv0[j];
        VTs[nxt][vd0 + 8 + j][vkk] = rv1[j];
      }
    }
    __syncthreads();
  }

  float* Og = O + ((size_t)b * Lq + m0 + wm * 32) * D + n0 + wn * 32;
#pragma unroll
  for (int r = 0; r < 8; ++r) {
    const int row = r + 8 * half;
    Og[(size_t)row * D + l16]             = c00[r];
    Og[(size_t)row * D + 16 + l16]        = c01[r];
    Og[(size_t)(row + 16) * D + l16]      = c10[r];
    Og[(size_t)(row + 16) * D + 16 + l16] = c11[r];
  }
}

// ===========================================================================
// FALLBACK PATH (convert-on-the-fly, round-2 kernels)
// ===========================================================================

__global__ __launch_bounds__(TB)
void qk_scores_kernel(const float* __restrict__ Q, const float* __restrict__ K,
                      float* __restrict__ S, int Lq, int Lk, int D) {
  __shared__ __bf16 Qs[2][BM][LDP];
  __shared__ __bf16 Ks[2][BN][LDP];

  const int b    = blockIdx.z;
  const int m0   = blockIdx.y * BM;
  const int n0   = blockIdx.x * BN;
  const int t    = threadIdx.x;
  const int lane = t & 31, wid = t >> 5;
  const int wm   = wid >> 1, wn = wid & 1;
  const int half = lane >> 4, l16 = lane & 15;

  const float* Ag = Q + ((size_t)b * Lq + m0) * D;
  const float* Bg = K + ((size_t)b * Lk + n0) * D;

  const int ar = t >> 1, ac0 = (t & 1) * 16;
  const int br = t >> 2, bc0 = (t & 3) * 8;
  const float* pa = Ag + (size_t)ar * D + ac0;
  const float* pb = Bg + (size_t)br * D + bc0;

  v8f c00 = {}, c01 = {}, c10 = {}, c11 = {};

  float4 a0 = *(const float4*)(pa + 0);
  float4 a1 = *(const float4*)(pa + 4);
  float4 a2 = *(const float4*)(pa + 8);
  float4 a3 = *(const float4*)(pa + 12);
  float4 b0 = *(const float4*)(pb + 0);
  float4 b1 = *(const float4*)(pb + 4);
  st4(&Qs[0][ar][ac0 + 0], a0); st4(&Qs[0][ar][ac0 + 4], a1);
  st4(&Qs[0][ar][ac0 + 8], a2); st4(&Qs[0][ar][ac0 + 12], a3);
  st4(&Ks[0][br][bc0 + 0], b0); st4(&Ks[0][br][bc0 + 4], b1);
  __syncthreads();

  const int NIT = D / BK;
  for (int it = 0; it < NIT; ++it) {
    const int cur = it & 1, nxt = cur ^ 1;
    const bool has = (it + 1 < NIT);
    if (has) {
      const int k0n = (it + 1) * BK;
      a0 = *(const float4*)(pa + k0n + 0);
      a1 = *(const float4*)(pa + k0n + 4);
      a2 = *(const float4*)(pa + k0n + 8);
      a3 = *(const float4*)(pa + k0n + 12);
      b0 = *(const float4*)(pb + k0n + 0);
      b1 = *(const float4*)(pb + k0n + 4);
    }

    Frag A0, A1, B0, B1;
    const __bf16* qa0 = &Qs[cur][wm * 32 + l16][0];
    const __bf16* qa1 = &Qs[cur][wm * 32 + 16 + l16][0];
    const __bf16* kb0 = &Ks[cur][wn * 32 + l16][0];
    const __bf16* kb1 = &Ks[cur][wn * 32 + 16 + l16][0];
    A0.h[0] = *(const v8bf*)(qa0 + 16 * half); A0.h[1] = *(const v8bf*)(qa0 + 16 * half + 8);
    A1.h[0] = *(const v8bf*)(qa1 + 16 * half); A1.h[1] = *(const v8bf*)(qa1 + 16 * half + 8);
    B0.h[0] = *(const v8bf*)(kb0 + 16 * half); B0.h[1] = *(const v8bf*)(kb0 + 16 * half + 8);
    B1.h[0] = *(const v8bf*)(kb1 + 16 * half); B1.h[1] = *(const v8bf*)(kb1 + 16 * half + 8);

    c00 = WMMA_BF16(A0.v, B0.v, c00);
    c01 = WMMA_BF16(A0.v, B1.v, c01);
    c10 = WMMA_BF16(A1.v, B0.v, c10);
    c11 = WMMA_BF16(A1.v, B1.v, c11);

    if (has) {
      st4(&Qs[nxt][ar][ac0 + 0], a0); st4(&Qs[nxt][ar][ac0 + 4], a1);
      st4(&Qs[nxt][ar][ac0 + 8], a2); st4(&Qs[nxt][ar][ac0 + 12], a3);
      st4(&Ks[nxt][br][bc0 + 0], b0); st4(&Ks[nxt][br][bc0 + 4], b1);
    }
    __syncthreads();
  }

  float* Sg = S + ((size_t)b * Lq + m0 + wm * 32) * Lk + n0 + wn * 32;
#pragma unroll
  for (int r = 0; r < 8; ++r) {
    const int row = r + 8 * half;
    Sg[(size_t)row * Lk + l16]             = c00[r];
    Sg[(size_t)row * Lk + 16 + l16]        = c01[r];
    Sg[(size_t)(row + 16) * Lk + l16]      = c10[r];
    Sg[(size_t)(row + 16) * Lk + 16 + l16] = c11[r];
  }
}

__global__ __launch_bounds__(TB)
void softmax_kernel(float* __restrict__ attn, int Lk) {
  __shared__ float red[TB];
  float* p = attn + (size_t)blockIdx.x * Lk;
  const int t = threadIdx.x;

  float m = -3.402823466e+38f;
  for (int i = t; i < Lk; i += TB) m = fmaxf(m, p[i]);
  red[t] = m; __syncthreads();
  for (int s = TB / 2; s > 0; s >>= 1) {
    if (t < s) red[t] = fmaxf(red[t], red[t + s]);
    __syncthreads();
  }
  m = red[0]; __syncthreads();

  float sum = 0.0f;
  for (int i = t; i < Lk; i += TB) {
    float e = __expf(p[i] - m);
    p[i] = e;
    sum += e;
  }
  red[t] = sum; __syncthreads();
  for (int s = TB / 2; s > 0; s >>= 1) {
    if (t < s) red[t] += red[t + s];
    __syncthreads();
  }
  const float inv = 1.0f / red[0];
  for (int i = t; i < Lk; i += TB) p[i] *= inv;
}

__global__ __launch_bounds__(TB)
void attn_v_kernel(const float* __restrict__ P, const float* __restrict__ V,
                   float* __restrict__ O, int Lq, int Lk, int D) {
  __shared__ __bf16 As[2][BM][LDP];
  __shared__ __bf16 VTs[2][BN][LDP];

  const int b    = blockIdx.z;
  const int m0   = blockIdx.y * BM;
  const int n0   = blockIdx.x * BN;
  const int t    = threadIdx.x;
  const int lane = t & 31, wid = t >> 5;
  const int wm   = wid >> 1, wn = wid & 1;
  const int half = lane >> 4, l16 = lane & 15;

  const float* Pg = P + ((size_t)b * Lq + m0) * Lk;
  const float* Vg = V + (size_t)b * Lk * D;

  const int ar  = t >> 1, ac0 = (t & 1) * 16;
  const int vkk = t >> 3, vd0 = (t & 7) * 8;
  const float* pa = Pg + (size_t)ar * Lk + ac0;

  v8f c00 = {}, c01 = {}, c10 = {}, c11 = {};

  float4 a0, a1, a2, a3, v0, v1;
  a0 = *(const float4*)(pa + 0);
  a1 = *(const float4*)(pa + 4);
  a2 = *(const float4*)(pa + 8);
  a3 = *(const float4*)(pa + 12);
  v0 = *(const float4*)(Vg + (size_t)vkk * D + n0 + vd0);
  v1 = *(const float4*)(Vg + (size_t)vkk * D + n0 + vd0 + 4);
  st4(&As[0][ar][ac0 + 0], a0); st4(&As[0][ar][ac0 + 4], a1);
  st4(&As[0][ar][ac0 + 8], a2); st4(&As[0][ar][ac0 + 12], a3);
  VTs[0][vd0 + 0][vkk] = f2bf(v0.x); VTs[0][vd0 + 1][vkk] = f2bf(v0.y);
  VTs[0][vd0 + 2][vkk] = f2bf(v0.z); VTs[0][vd0 + 3][vkk] = f2bf(v0.w);
  VTs[0][vd0 + 4][vkk] = f2bf(v1.x); VTs[0][vd0 + 5][vkk] = f2bf(v1.y);
  VTs[0][vd0 + 6][vkk] = f2bf(v1.z); VTs[0][vd0 + 7][vkk] = f2bf(v1.w);
  __syncthreads();

  const int NIT = Lk / BK;
  for (int it = 0; it < NIT; ++it) {
    const int cur = it & 1, nxt = cur ^ 1;
    const bool has = (it + 1 < NIT);
    if (has) {
      const int k0n = (it + 1) * BK;
      a0 = *(const float4*)(pa + k0n + 0);
      a1 = *(const float4*)(pa + k0n + 4);
      a2 = *(const float4*)(pa + k0n + 8);
      a3 = *(const float4*)(pa + k0n + 12);
      v0 = *(const float4*)(Vg + (size_t)(k0n + vkk) * D + n0 + vd0);
      v1 = *(const float4*)(Vg + (size_t)(k0n + vkk) * D + n0 + vd0 + 4);
    }

    Frag A0, A1, B0, B1;
    const __bf16* pa0 = &As[cur][wm * 32 + l16][0];
    const __bf16* pa1 = &As[cur][wm * 32 + 16 + l16][0];
    const __bf16* pb0 = &VTs[cur][wn * 32 + l16][0];
    const __bf16* pb1 = &VTs[cur][wn * 32 + 16 + l16][0];
    A0.h[0] = *(const v8bf*)(pa0 + 16 * half); A0.h[1] = *(const v8bf*)(pa0 + 16 * half + 8);
    A1.h[0] = *(const v8bf*)(pa1 + 16 * half); A1.h[1] = *(const v8bf*)(pa1 + 16 * half + 8);
    B0.h[0] = *(const v8bf*)(pb0 + 16 * half); B0.h[1] = *(const v8bf*)(pb0 + 16 * half + 8);
    B1.h[0] = *(const v8bf*)(pb1 + 16 * half); B1.h[1] = *(const v8bf*)(pb1 + 16 * half + 8);

    c00 = WMMA_BF16(A0.v, B0.v, c00);
    c01 = WMMA_BF16(A0.v, B1.v, c01);
    c10 = WMMA_BF16(A1.v, B0.v, c10);
    c11 = WMMA_BF16(A1.v, B1.v, c11);

    if (has) {
      st4(&As[nxt][ar][ac0 + 0], a0); st4(&As[nxt][ar][ac0 + 4], a1);
      st4(&As[nxt][ar][ac0 + 8], a2); st4(&As[nxt][ar][ac0 + 12], a3);
      VTs[nxt][vd0 + 0][vkk] = f2bf(v0.x); VTs[nxt][vd0 + 1][vkk] = f2bf(v0.y);
      VTs[nxt][vd0 + 2][vkk] = f2bf(v0.z); VTs[nxt][vd0 + 3][vkk] = f2bf(v0.w);
      VTs[nxt][vd0 + 4][vkk] = f2bf(v1.x); VTs[nxt][vd0 + 5][vkk] = f2bf(v1.y);
      VTs[nxt][vd0 + 6][vkk] = f2bf(v1.z); VTs[nxt][vd0 + 7][vkk] = f2bf(v1.w);
    }
    __syncthreads();
  }

  float* Og = O + ((size_t)b * Lq + m0 + wm * 32) * D + n0 + wn * 32;
#pragma unroll
  for (int r = 0; r < 8; ++r) {
    const int row = r + 8 * half;
    Og[(size_t)row * D + l16]             = c00[r];
    Og[(size_t)row * D + 16 + l16]        = c01[r];
    Og[(size_t)(row + 16) * D + l16]      = c10[r];
    Og[(size_t)(row + 16) * D + 16 + l16] = c11[r];
  }
}

// ---------------------------------------------------------------------------
extern "C" void kernel_launch(void* const* d_in, const int* in_sizes, int n_in,
                              void* d_out, int out_size, void* d_ws, size_t ws_size,
                              hipStream_t stream) {
  (void)in_sizes; (void)n_in; (void)out_size;
  const int B = 8, Lq = 2048, Lk = 2048, D = 1024;

  const float* Q  = (const float*)d_in[0];   // "output" in reference = query
  const float* KV = (const float*)d_in[1];   // "inputs" = key = value

  float* ctx  = (float*)d_out;                 // (B, Lq, D)
  float* attn = ctx + (size_t)B * Lq * D;      // (B, Lq, Lk)

  dim3 blk(TB);
  dim3 g1(Lk / BN, Lq / BM, B);
  dim3 g3(D / BN, Lq / BM, B);

  const size_t nQ = (size_t)B * Lq * D;
  const size_t nK = (size_t)B * Lk * D;
  const size_t nP = (size_t)B * Lq * Lk;
  const size_t need = (nQ + nK + nP) * sizeof(__bf16);

  if (d_ws != nullptr && ws_size >= need) {
    // ---- fast path: one-shot bf16 conversion, pure-copy WMMA GEMMs ----
    __bf16* Qb = (__bf16*)d_ws;
    __bf16* Kb = Qb + nQ;           // also serves as V (same tensor)
    __bf16* Pb = Kb + nK;

    f32_to_bf16_kernel<<<dim3(4096), blk, 0, stream>>>(Q,  Qb, (long long)(nQ / 4));
    f32_to_bf16_kernel<<<dim3(4096), blk, 0, stream>>>(KV, Kb, (long long)(nK / 4));

    qk_scores_bf_kernel<<<g1, blk, 0, stream>>>(Qb, Kb, attn, Lq, Lk, D);
    softmax_bf_kernel<<<dim3(B * Lq), blk, 0, stream>>>(attn, Pb, Lk);
    attn_v_bf_kernel<<<g3, blk, 0, stream>>>(Pb, Kb, ctx, Lq, Lk, D);
  } else {
    // ---- fallback: convert-on-the-fly inside the GEMMs ----
    qk_scores_kernel<<<g1, blk, 0, stream>>>(Q, KV, attn, Lq, Lk, D);
    softmax_kernel<<<dim3(B * Lq), blk, 0, stream>>>(attn, Lk);
    attn_v_kernel<<<g3, blk, 0, stream>>>(attn, KV, ctx, Lq, Lk, D);
  }
}
